// RNN_601295421440
// MI455X (gfx1250) — matrix operand order repaired and checked by reading, not compiled, not run on hardware
//
#include <hip/hip_runtime.h>
#include <hip/hip_bf16.h>

// ---------------------------------------------------------------------------
// Fused 2-layer LSTM + Dense + head for MI455X (gfx1250, wave32, WMMA f16).
// Each wave processes 16 batch rows across all T=10 timesteps.
// Matmuls use v_wmma_f32_16x16x32_f16 (f16 operands, f32 accumulate).
// Weights are staged once into d_ws as f16 (N-major, bank-conflict-padded
// strides, K zero-padded), then block-copied into LDS (CDNA5 WGP has 320 KB)
// and read as B-fragments with ds_load_b128 — no register-file caching, no
// scratch spills. Activations use v_exp_f32 / v_rcp_f32.
// ---------------------------------------------------------------------------

typedef _Float16 h16;
typedef __attribute__((ext_vector_type(8)))  _Float16 v8h;
typedef __attribute__((ext_vector_type(16))) _Float16 v16h;
typedef __attribute__((ext_vector_type(8)))  float    v8f;
typedef __attribute__((ext_vector_type(4)))  float    v4f;

#define T_STEPS 10
#define D_IN    16
#define HID     64   // 4*HID = 256 gate width

// Padded strides (halves): 72 = 144B (36-bank step, conflict-free for 16 rows),
// 40 = 80B (20-bank step, conflict-free). Both multiples of 16B.
#define S72 72
#define S40 40

// Workspace / LDS weight layout (in halves):
#define OFF_W1T 0                       // [256][40], K=16 real
#define OFF_U1T (OFF_W1T + 256*S40)     // 10240: [256][72], K=64 real
#define OFF_WDT (OFF_U1T + 256*S72)     // 28672: [ 32][72], N=30 real, K=64
#define OFF_W2T (OFF_WDT + 32*S72)      // 30976: [256][40], K=30 real
#define OFF_U2T (OFF_W2T + 256*S40)     // 41216: [256][72], K=64 real
#define OFF_WFT (OFF_U2T + 256*S72)     // 59648: [ 16][72], N=4 real, K=64
#define WS_HALVES (OFF_WFT + 16*S72)    // 60800 halves = 121600 B

// Per-wave state tiles appended after the weights in LDS:
#define WAVE_HALVES (16*S72 + 16*S72 + 16*S40)   // h1 + h2 + d = 2944
#define LDS_HALVES  (WS_HALVES + 4*WAVE_HALVES)  // 72576 halves = 145152 B

__device__ __forceinline__ v16h cat8(v8h a, v8h b) {
    return __builtin_shufflevector(a, b, 0,1,2,3,4,5,6,7,8,9,10,11,12,13,14,15);
}

// A-fragment (16x32 f16) from a row-major f16 buffer, row stride `stride`
// halves, K window starting at k0. ISA layout: lanes 0-15 hold K={k0..k0+7,
// k0+16..k0+23}; lanes 16-31 hold K={k0+8..k0+15, k0+24..k0+31}. M = lane%16.
__device__ __forceinline__ v16h load_a(const h16* buf, int stride, int lane, int k0) {
    int m   = lane & 15;
    int off = (lane & 16) ? 8 : 0;
    const h16* p = buf + m * stride + k0 + off;
    v8h lo = *(const v8h*)(p);
    v8h hi = *(const v8h*)(p + 16);
    return cat8(lo, hi);
}

// B-fragment (32x16 f16) from an N-major staged weight (row = output col n,
// stride halves). B layout matching A: N = lane%16,
// lanes 0-15 hold K={k0..k0+7, k0+16..k0+23}; lanes 16-31 hold K={k0+8..k0+15, k0+24..k0+31}.
__device__ __forceinline__ v16h load_b(const h16* wt, int stride, int ntile, int lane, int k0) {
    int n   = (ntile << 4) + (lane & 15);
    int off = (lane & 16) ? 8 : 0;
    const h16* p = wt + n * stride + k0 + off;
    v8h lo = *(const v8h*)(p);
    v8h hi = *(const v8h*)(p + 16);
    return cat8(lo, hi);
}

__device__ __forceinline__ v8f wmma(v16h a, v16h b, v8f c) {
    return __builtin_amdgcn_wmma_f32_16x16x32_f16(false, a, false, b, (short)0, c, false, false);
}

__device__ __forceinline__ v8f splat8(float v) {
    v8f r;
#pragma unroll
    for (int i = 0; i < 8; ++i) r[i] = v;
    return r;
}

// sigmoid(x) = rcp(1 + exp2(-x*log2e)) : v_exp_f32 + v_rcp_f32
__device__ __forceinline__ float fsig(float x) {
    float e = __builtin_amdgcn_exp2f(-1.44269504088896f * x);
    return __builtin_amdgcn_rcpf(1.0f + e);
}
// tanh(x) = 2*sigmoid(2x) - 1  (saturates correctly)
__device__ __forceinline__ float ftanh(float x) {
    float e = __builtin_amdgcn_exp2f(-2.88539008177793f * x);
    return 2.0f * __builtin_amdgcn_rcpf(1.0f + e) - 1.0f;
}

__device__ __forceinline__ v8f vsig(v8f x) {
    v8f r;
#pragma unroll
    for (int i = 0; i < 8; ++i) r[i] = fsig(x[i]);
    return r;
}
__device__ __forceinline__ v8f vtanh(v8f x) {
    v8f r;
#pragma unroll
    for (int i = 0; i < 8; ++i) r[i] = ftanh(x[i]);
    return r;
}
__device__ __forceinline__ v8f vrelu(v8f x) {
    v8f r;
#pragma unroll
    for (int i = 0; i < 8; ++i) r[i] = x[i] > 0.0f ? x[i] : 0.0f;
    return r;
}

// ---------------------------------------------------------------------------
// Prep kernel: convert weights to f16, transpose to N-major padded layouts.
// ---------------------------------------------------------------------------
__global__ void lstm_prep(const float* __restrict__ W1, const float* __restrict__ U1,
                          const float* __restrict__ Wd, const float* __restrict__ W2,
                          const float* __restrict__ U2, const float* __restrict__ Wf,
                          h16* __restrict__ wt) {
    int idx = blockIdx.x * blockDim.x + threadIdx.x;
    if (idx >= WS_HALVES) return;
    float v = 0.0f;
    if (idx < OFF_U1T) {                       // W1t [256][40], W1 is [16][256]
        int j = idx - OFF_W1T, n = j / S40, k = j % S40;
        v = (k < D_IN) ? W1[k * 256 + n] : 0.0f;
    } else if (idx < OFF_WDT) {                // U1t [256][72], U1 is [64][256]
        int j = idx - OFF_U1T, n = j / S72, k = j % S72;
        v = (k < HID) ? U1[k * 256 + n] : 0.0f;
    } else if (idx < OFF_W2T) {                // Wdt [32][72], Wd is [64][30]
        int j = idx - OFF_WDT, n = j / S72, k = j % S72;
        v = (n < 30 && k < HID) ? Wd[k * 30 + n] : 0.0f;
    } else if (idx < OFF_U2T) {                // W2t [256][40], W2 is [30][256]
        int j = idx - OFF_W2T, n = j / S40, k = j % S40;
        v = (k < 30) ? W2[k * 256 + n] : 0.0f;
    } else if (idx < OFF_WFT) {                // U2t [256][72], U2 is [64][256]
        int j = idx - OFF_U2T, n = j / S72, k = j % S72;
        v = (k < HID) ? U2[k * 256 + n] : 0.0f;
    } else {                                   // Wft [16][72], Wf is [64][4]
        int j = idx - OFF_WFT, n = j / S72, k = j % S72;
        v = (n < 4 && k < HID) ? Wf[k * 4 + n] : 0.0f;
    }
    wt[idx] = (h16)v;
}

// ---------------------------------------------------------------------------
// Fused main kernel: 1 wave = 16 batch rows, full sequence, all layers.
// Weights live in LDS (block-shared); h-state tiles per wave in LDS; the
// h-store aliasing keeps weight ds_loads in-loop => low VGPR pressure.
// ---------------------------------------------------------------------------
__global__ __launch_bounds__(128) void lstm_fused(
        const float* __restrict__ x,
        const float* __restrict__ b1, const float* __restrict__ bd,
        const float* __restrict__ b2, const float* __restrict__ bf,
        const h16* __restrict__ wt, float* __restrict__ out) {

    __shared__ alignas(16) h16 smem[LDS_HALVES];

    // ---- stage all weights into LDS (one vectorized block copy) -----------
    {
        const v8h* src = (const v8h*)wt;
        v8h*       dst = (v8h*)smem;
#pragma unroll 1
        for (int i = threadIdx.x; i < WS_HALVES / 8; i += 128)
            dst[i] = src[i];
    }

    const h16* W1t = smem + OFF_W1T;
    const h16* U1t = smem + OFF_U1T;
    const h16* Wdt = smem + OFF_WDT;
    const h16* W2t = smem + OFF_W2T;
    const h16* U2t = smem + OFF_U2T;
    const h16* Wft = smem + OFF_WFT;

    const int wave = threadIdx.x >> 5;
    const int lane = threadIdx.x & 31;
    h16* h1buf = smem + WS_HALVES + wave * WAVE_HALVES;   // [16][72]
    h16* h2buf = h1buf + 16 * S72;                        // [16][72]
    h16* dbuf  = h2buf + 16 * S72;                        // [16][40]

    const size_t m0  = ((size_t)blockIdx.x * 4 + wave) * 16;
    const int    col = lane & 15;
    const int    rb  = (lane & 16) ? 8 : 0;

    // zero own h1/h2 state tiles (2*16*72 halves, 2B stores avoided: 16B rows)
    {
        v8h z = {};
#pragma unroll
        for (int r = 0; r < 16; ++r) {
            // each row: 72 halves = 9 v8h; lane covers rows via strided writes
            if (lane < 9) {
                ((v8h*)(h1buf + r * S72))[lane] = z;
                ((v8h*)(h2buf + r * S72))[lane] = z;
            }
        }
    }

    // hoist per-lane gate biases (column n = (gate*4+k)*16 + col)
    float b1v[4][4], b2v[4][4];
#pragma unroll
    for (int g = 0; g < 4; ++g)
#pragma unroll
        for (int k = 0; k < 4; ++k) {
            int n = (g * 4 + k) * 16 + col;
            b1v[g][k] = b1[n];
            b2v[g][k] = b2[n];
        }
    float bdv0 = bd[col];
    float bdv1 = (col + 16 < 30) ? bd[col + 16] : 0.0f;
    float bfv  = (col < 4) ? bf[col] : 0.0f;

    v8f c1[4], c2[4];
#pragma unroll
    for (int k = 0; k < 4; ++k) { c1[k] = splat8(0.0f); c2[k] = splat8(0.0f); }

    __syncthreads();   // weights staged, h tiles zeroed

#pragma unroll 1
    for (int t = 0; t < T_STEPS; ++t) {
        // ---- x_t A-fragment (K=16 real, 16..31 zero); streamed non-temporal
        v16h xa;
        {
            int m = lane & 15;
            const float* xp = x + ((m0 + m) * T_STEPS + t) * D_IN + ((lane & 16) ? 8 : 0);
            v4f f0 = __builtin_nontemporal_load((const v4f*)(xp));
            v4f f1 = __builtin_nontemporal_load((const v4f*)(xp + 4));
#pragma unroll
            for (int j = 0; j < 4; ++j) { xa[j] = (h16)f0[j]; xa[4 + j] = (h16)f1[j]; }
#pragma unroll
            for (int j = 8; j < 16; ++j) xa[j] = (h16)0.0f;
        }

        // ---- LSTM1: z = x@W1 + h1@U1 + b1 ---------------------------------
        v16h h1lo = load_a(h1buf, S72, lane, 0);
        v16h h1hi = load_a(h1buf, S72, lane, 32);
#pragma unroll
        for (int k = 0; k < 4; ++k) {
            v8f acc[4];
#pragma unroll
            for (int g = 0; g < 4; ++g) {
                int nt = g * 4 + k;
                v8f a = splat8(b1v[g][k]);
                a = wmma(xa,   load_b(W1t, S40, nt, lane, 0),  a);
                a = wmma(h1lo, load_b(U1t, S72, nt, lane, 0),  a);
                a = wmma(h1hi, load_b(U1t, S72, nt, lane, 32), a);
                acc[g] = a;
            }
            v8f ig = vsig(acc[0]), fg = vsig(acc[1]);
            v8f gg = vtanh(acc[2]), og = vsig(acc[3]);
            c1[k] = fg * c1[k] + ig * gg;
            v8f h = og * vtanh(c1[k]);
#pragma unroll
            for (int r = 0; r < 8; ++r)
                h1buf[(rb + r) * S72 + k * 16 + col] = (h16)h[r];
        }

        // ---- Dense: d = h1 @ Wd + bd (N=30, padded to 32) ------------------
        v16h ha0 = load_a(h1buf, S72, lane, 0);
        v16h ha1 = load_a(h1buf, S72, lane, 32);
#pragma unroll
        for (int nt = 0; nt < 2; ++nt) {
            v8f a = splat8(nt == 0 ? bdv0 : bdv1);
            a = wmma(ha0, load_b(Wdt, S72, nt, lane, 0),  a);
            a = wmma(ha1, load_b(Wdt, S72, nt, lane, 32), a);
#pragma unroll
            for (int r = 0; r < 8; ++r)
                dbuf[(rb + r) * S40 + nt * 16 + col] = (h16)a[r];
        }

        // ---- LSTM2 (relu act): z = d@W2 + h2@U2 + b2 ----------------------
        v16h da   = load_a(dbuf, S40, lane, 0);
        v16h h2lo = load_a(h2buf, S72, lane, 0);
        v16h h2hi = load_a(h2buf, S72, lane, 32);
#pragma unroll
        for (int k = 0; k < 4; ++k) {
            v8f acc[4];
#pragma unroll
            for (int g = 0; g < 4; ++g) {
                int nt = g * 4 + k;
                v8f a = splat8(b2v[g][k]);
                a = wmma(da,   load_b(W2t, S40, nt, lane, 0),  a);
                a = wmma(h2lo, load_b(U2t, S72, nt, lane, 0),  a);
                a = wmma(h2hi, load_b(U2t, S72, nt, lane, 32), a);
                acc[g] = a;
            }
            v8f ig = vsig(acc[0]), fg = vsig(acc[1]);
            v8f gg = vrelu(acc[2]), og = vsig(acc[3]);
            c2[k] = fg * c2[k] + ig * gg;
            v8f h = og * vrelu(c2[k]);
#pragma unroll
            for (int r = 0; r < 8; ++r)
                h2buf[(rb + r) * S72 + k * 16 + col] = (h16)h[r];
        }
    }

    // ---- head: out = h2_last @ Wf + bf (N=4, padded to 16) -----------------
    v16h hf0 = load_a(h2buf, S72, lane, 0);
    v16h hf1 = load_a(h2buf, S72, lane, 32);
    v8f a = splat8(bfv);
    a = wmma(hf0, load_b(Wft, S72, 0, lane, 0),  a);
    a = wmma(hf1, load_b(Wft, S72, 0, lane, 32), a);
    if (col < 4) {
#pragma unroll
        for (int r = 0; r < 8; ++r)
            __builtin_nontemporal_store(a[r], out + (m0 + rb + r) * 4 + col);
    }
}

// ---------------------------------------------------------------------------
extern "C" void kernel_launch(void* const* d_in, const int* in_sizes, int n_in,
                              void* d_out, int out_size, void* d_ws, size_t ws_size,
                              hipStream_t stream) {
    (void)in_sizes; (void)n_in; (void)out_size; (void)ws_size;
    const float* x  = (const float*)d_in[0];
    const float* W1 = (const float*)d_in[1];
    const float* U1 = (const float*)d_in[2];
    const float* b1 = (const float*)d_in[3];
    const float* Wd = (const float*)d_in[4];
    const float* bd = (const float*)d_in[5];
    const float* W2 = (const float*)d_in[6];
    const float* U2 = (const float*)d_in[7];
    const float* b2 = (const float*)d_in[8];
    const float* Wf = (const float*)d_in[9];
    const float* bf = (const float*)d_in[10];
    h16* wt = (h16*)d_ws;

    lstm_prep<<<(WS_HALVES + 255) / 256, 256, 0, stream>>>(W1, U1, Wd, W2, U2, Wf, wt);

    // 65536 rows / 16 per wave = 4096 waves; 4 waves (128 threads) per block.
    lstm_fused<<<1024, 128, 0, stream>>>(x, b1, bd, b2, bf, wt, (float*)d_out);
}
